// ResidualAttentionBlock_14731737825916
// MI455X (gfx1250) — matrix operand order repaired and checked by reading, not compiled, not run on hardware
//
#include <hip/hip_runtime.h>
#include <hip/hip_bf16.h>

// ---------------------------------------------------------------------------
// Residual attention block (CLIP), D=1024, H=16, seq=1024, batch=8, fp32 I/O.
// GEMMs + attention on v_wmma_f32_16x16x32_bf16; async-LDS V staging; DPP
// ROW_XMASK softmax reductions (no ds_bpermute in the hot loop).
// ---------------------------------------------------------------------------

#define D_MODEL 1024
#define SEQ     1024
#define NB      8
#define NH      16
#define HC      64              // head channels
#define NTOK    (SEQ * NB)      // 8192 tokens

typedef __attribute__((ext_vector_type(16))) __bf16 v16bf;
typedef __attribute__((ext_vector_type(8)))  float  v8f;

union Frag {                    // one 16-lane-striped bf16 WMMA operand
    v16bf v;
    uint4 q[2];
    unsigned short us[16];
};

__device__ __forceinline__ unsigned short f2bf(float f) {
    unsigned int u = __float_as_uint(f);
    u += 0x7FFFu + ((u >> 16) & 1u);      // round-to-nearest-even
    return (unsigned short)(u >> 16);
}

__device__ __forceinline__ v8f wmma_bf16(v16bf a, v16bf b, v8f c) {
    return __builtin_amdgcn_wmma_f32_16x16x32_bf16(
        false, a, false, b, (short)0, c, false, false);
}

// lane ^ MASK within a 16-lane DPP row (ROW_XMASK = 0x160 | mask), VALU-only
template <int MASK>
__device__ __forceinline__ float dpp_xor(float x) {
    int i = __float_as_int(x);
    int r = __builtin_amdgcn_update_dpp(i, i, 0x160 + MASK, 0xf, 0xf, true);
    return __int_as_float(r);
}

// ---------------------------------------------------------------------------
// fp32 -> bf16 weight conversion (weights then stay L2-resident, ~25 MB)
// ---------------------------------------------------------------------------
__global__ void __launch_bounds__(256)
cvt_f32_bf16(const float* __restrict__ in, unsigned short* __restrict__ out, int n) {
    int i = blockIdx.x * blockDim.x + threadIdx.x;
    if (i < n) out[i] = f2bf(in[i]);
}

// ---------------------------------------------------------------------------
// LayerNorm over D=1024, fp32 in -> bf16 out (token-major [n, s, d]).
// PERM_IN: input is [s, n, d] (the original x); else token-major.
// ---------------------------------------------------------------------------
template <bool PERM_IN>
__global__ void __launch_bounds__(256)
ln_kernel(const float* __restrict__ in, const float* __restrict__ w,
          const float* __restrict__ b, unsigned short* __restrict__ out) {
    const int t  = blockIdx.x;              // token, batch-major: t = nI*SEQ + sI
    const int nI = t >> 10;
    const int sI = t & (SEQ - 1);
    const size_t inRow = PERM_IN ? (size_t)(sI * NB + nI) : (size_t)t;
    const float* xr = in + inRow * D_MODEL;

    const int tid = threadIdx.x;
    const int d   = tid * 4;
    float4 xv = *(const float4*)(xr + d);

    float s  = xv.x + xv.y + xv.z + xv.w;
    float ss = xv.x * xv.x + xv.y * xv.y + xv.z * xv.z + xv.w * xv.w;
    #pragma unroll
    for (int m = 16; m >= 1; m >>= 1) {
        s  += __shfl_xor(s,  m);
        ss += __shfl_xor(ss, m);
    }
    __shared__ float red[16];
    const int wv = tid >> 5, ln = tid & 31;
    if (ln == 0) { red[wv] = s; red[8 + wv] = ss; }
    __syncthreads();
    float ts = 0.f, tss = 0.f;
    #pragma unroll
    for (int i = 0; i < 8; ++i) { ts += red[i]; tss += red[8 + i]; }

    const float mu  = ts * (1.0f / D_MODEL);
    const float var = tss * (1.0f / D_MODEL) - mu * mu;
    const float rs  = rsqrtf(var + 1e-5f);

    float4 wv4 = *(const float4*)(w + d);
    float4 bv4 = *(const float4*)(b + d);
    unsigned short* orow = out + (size_t)t * D_MODEL;
    orow[d + 0] = f2bf((xv.x - mu) * rs * wv4.x + bv4.x);
    orow[d + 1] = f2bf((xv.y - mu) * rs * wv4.y + bv4.y);
    orow[d + 2] = f2bf((xv.z - mu) * rs * wv4.z + bv4.z);
    orow[d + 3] = f2bf((xv.w - mu) * rs * wv4.w + bv4.w);
}

// ---------------------------------------------------------------------------
// GEMM: C[M,N] = A[M,K] * W[N,K]^T + bias, bf16 in, fp32 accumulate.
// 256 threads = 8 waves arranged 2(M) x 4(N); WG tile 128x128;
// wave tile 64x32 = 4x2 WMMA (8 WMMA : 12 b128 loads per k-step).
// MODE 0: bf16 out      MODE 1: bf16 out + QuickGELU
// MODE 2: fp32 out = v + res permuted([s,n,d]->token)   (residual 1)
// MODE 3: fp32 out permuted (token->[s,n,d]) = v + res  (residual 2, d_out)
// ---------------------------------------------------------------------------
template <int MODE>
__global__ void __launch_bounds__(256)
gemm_bf16(const unsigned short* __restrict__ A,
          const unsigned short* __restrict__ W,
          const float* __restrict__ bias,
          const float* __restrict__ res,
          void* __restrict__ outP,
          int M, int N, int K) {
    const int lane = threadIdx.x & 31;
    const int wave = threadIdx.x >> 5;
    const int l15  = lane & 15;
    const int hi   = lane >> 4;
    const int mW = blockIdx.y * 128 + (wave >> 2) * 64;   // 2 waves along M
    const int nW = blockIdx.x * 128 + (wave & 3) * 32;    // 4 waves along N

    v8f acc[4][2];
    #pragma unroll
    for (int i = 0; i < 4; ++i)
        #pragma unroll
        for (int j = 0; j < 2; ++j)
            acc[i][j] = (v8f){0.f, 0.f, 0.f, 0.f, 0.f, 0.f, 0.f, 0.f};

    const unsigned short* aBase[4];
    const unsigned short* wBase[2];
    #pragma unroll
    for (int mi = 0; mi < 4; ++mi)
        aBase[mi] = A + (size_t)(mW + mi * 16 + l15) * K + hi * 16;
    #pragma unroll
    for (int ni = 0; ni < 2; ++ni)
        wBase[ni] = W + (size_t)(nW + ni * 16 + l15) * K + hi * 16;

    for (int k = 0; k < K; k += 32) {
        __builtin_prefetch(aBase[0] + k + 64, 0, 3);   // global_prefetch_b8
        __builtin_prefetch(wBase[0] + k + 64, 0, 3);
        Frag a[4], b[2];
        #pragma unroll
        for (int mi = 0; mi < 4; ++mi) {
            a[mi].q[0] = *(const uint4*)(aBase[mi] + k);
            a[mi].q[1] = *(const uint4*)(aBase[mi] + k + 8);
        }
        #pragma unroll
        for (int ni = 0; ni < 2; ++ni) {
            b[ni].q[0] = *(const uint4*)(wBase[ni] + k);
            b[ni].q[1] = *(const uint4*)(wBase[ni] + k + 8);
        }
        #pragma unroll
        for (int mi = 0; mi < 4; ++mi)
            #pragma unroll
            for (int ni = 0; ni < 2; ++ni)
                acc[mi][ni] = wmma_bf16(a[mi].v, b[ni].v, acc[mi][ni]);
    }

    // Epilogue. C layout: lane l: n = l&15; VGPR r -> m = r + 8*(l>=16).
    #pragma unroll
    for (int mi = 0; mi < 4; ++mi) {
        #pragma unroll
        for (int ni = 0; ni < 2; ++ni) {
            const int n  = nW + ni * 16 + l15;
            const float bv = bias[n];
            #pragma unroll
            for (int r = 0; r < 8; ++r) {
                const int m = mW + mi * 16 + r + 8 * hi;
                float v = acc[mi][ni][r] + bv;
                if (MODE == 0) {
                    ((unsigned short*)outP)[(size_t)m * N + n] = f2bf(v);
                } else if (MODE == 1) {
                    v = v / (1.0f + __expf(-1.702f * v));   // QuickGELU
                    ((unsigned short*)outP)[(size_t)m * N + n] = f2bf(v);
                } else if (MODE == 2) {
                    const int nI = m >> 10, sI = m & (SEQ - 1);
                    v += res[(size_t)(sI * NB + nI) * D_MODEL + n];
                    ((float*)outP)[(size_t)m * N + n] = v;
                } else {
                    v += res[(size_t)m * N + n];
                    const int nI = m >> 10, sI = m & (SEQ - 1);
                    ((float*)outP)[(size_t)(sI * NB + nI) * D_MODEL + n] = v;
                }
            }
        }
    }
}

// ---------------------------------------------------------------------------
// Flash attention: 1 WG per (batch, head, 64 q rows); 4 waves x 16 q rows.
// V tile copied raw to LDS with GLOBAL_LOAD_ASYNC_TO_LDS_B128 (ASYNCcnt),
// softmax row-reductions via DPP ROW_XMASK (pure VALU), P re-laid out C->A
// through per-wave LDS, P@V B-frags gathered transposed from the raw V tile.
// ---------------------------------------------------------------------------
__global__ void __launch_bounds__(128)
attn_kernel(const unsigned short* __restrict__ qkv,
            unsigned short* __restrict__ outMix) {
    const int qTile = blockIdx.x;   // 0..15  (64 q rows each)
    const int head  = blockIdx.y;   // 0..15
    const int batch = blockIdx.z;   // 0..7
    const int tid   = threadIdx.x;
    const int lane  = tid & 31, wave = tid >> 5;
    const int l15   = lane & 15, hi = lane >> 4;

    __shared__ unsigned short Vr[32][HC];        // raw V tile [key][c]
    __shared__ unsigned short Pl[4][16][32];     // per-wave P tile [m][key]

    const size_t rowStride = 3 * D_MODEL;
    const size_t base = (size_t)batch * SEQ * rowStride;

    // Q A-fragments, held in registers for the whole kernel
    const int qRow = qTile * 64 + wave * 16 + l15;
    const unsigned short* qp = qkv + base + (size_t)qRow * rowStride + head * HC;
    Frag qa[2];
    #pragma unroll
    for (int cp = 0; cp < 2; ++cp) {
        qa[cp].q[0] = *(const uint4*)(qp + cp * 32 + hi * 16);
        qa[cp].q[1] = *(const uint4*)(qp + cp * 32 + hi * 16 + 8);
    }

    v8f o[4];
    #pragma unroll
    for (int cg = 0; cg < 4; ++cg)
        o[cg] = (v8f){0.f, 0.f, 0.f, 0.f, 0.f, 0.f, 0.f, 0.f};
    float rm[8], rl[8];
    #pragma unroll
    for (int r = 0; r < 8; ++r) { rm[r] = -1e30f; rl[r] = 0.f; }

    for (int kb = 0; kb < SEQ; kb += 32) {
        __syncthreads();                 // previous iteration done with Vr
        {   // async-copy V tile (32 keys x 64 c, 4 KB) raw into LDS:
            // 256 x 16B chunks over 128 threads (2 per thread)
            #pragma unroll
            for (int i = 0; i < 2; ++i) {
                const int ch  = tid * 2 + i;         // 0..255
                const int key = ch >> 3;             // 0..31
                const int c0  = (ch & 7) * 8;        // 0..56
                const unsigned short* vp = qkv + base + (size_t)(kb + key) * rowStride
                                           + 2 * D_MODEL + head * HC + c0;
                unsigned ldsa = (unsigned)(size_t)&Vr[key][c0];
                asm volatile("global_load_async_to_lds_b128 %0, %1, off"
                             :: "v"(ldsa), "v"(vp) : "memory");
            }
            asm volatile("s_wait_asynccnt 0x0" ::: "memory");
        }
        __syncthreads();

        // S tiles (16 q x 16 keys each), contraction over c=64 in 2 chunks
        v8f s0 = (v8f){0.f, 0.f, 0.f, 0.f, 0.f, 0.f, 0.f, 0.f};
        v8f s1 = s0;
        #pragma unroll
        for (int cp = 0; cp < 2; ++cp) {
            const unsigned short* kp0 = qkv + base + (size_t)(kb + l15) * rowStride
                                        + D_MODEL + head * HC + cp * 32 + hi * 16;
            const unsigned short* kp1 = kp0 + 16 * rowStride;
            Frag b0, b1;
            b0.q[0] = *(const uint4*)(kp0); b0.q[1] = *(const uint4*)(kp0 + 8);
            b1.q[0] = *(const uint4*)(kp1); b1.q[1] = *(const uint4*)(kp1 + 8);
            s0 = wmma_bf16(qa[cp].v, b0.v, s0);
            s1 = wmma_bf16(qa[cp].v, b1.v, s1);
        }

        // online softmax over these 32 keys; keys live across lanes in the
        // C layout, so row reductions are XOR masks 1/2/4/8 -> DPP ROW_XMASK
        #pragma unroll
        for (int r = 0; r < 8; ++r) {
            float v0 = s0[r] * 0.125f;           // 1/sqrt(64)
            float v1 = s1[r] * 0.125f;
            float mx = fmaxf(v0, v1);
            mx = fmaxf(mx, dpp_xor<1>(mx));
            mx = fmaxf(mx, dpp_xor<2>(mx));
            mx = fmaxf(mx, dpp_xor<4>(mx));
            mx = fmaxf(mx, dpp_xor<8>(mx));
            const float mnew = fmaxf(rm[r], mx);
            const float corr = __expf(rm[r] - mnew);
            rm[r] = mnew;
            const float p0 = __expf(v0 - mnew);
            const float p1 = __expf(v1 - mnew);
            float ps = p0 + p1;
            ps += dpp_xor<1>(ps);
            ps += dpp_xor<2>(ps);
            ps += dpp_xor<4>(ps);
            ps += dpp_xor<8>(ps);
            rl[r] = rl[r] * corr + ps;
            o[0][r] *= corr; o[1][r] *= corr; o[2][r] *= corr; o[3][r] *= corr;
            const int mrow = r + 8 * hi;
            Pl[wave][mrow][l15]      = f2bf(p0);
            Pl[wave][mrow][16 + l15] = f2bf(p1);
        }
        __builtin_amdgcn_wave_barrier();   // keep DS stores before DS reads

        // P as A-fragment (16 q x 32 keys), read back in A layout
        Frag pa;
        pa.q[0] = *(const uint4*)&Pl[wave][l15][hi * 16];
        pa.q[1] = *(const uint4*)&Pl[wave][l15][hi * 16 + 8];

        // O += P @ V ; B-frags gathered transposed from raw V tile:
        // lane: c = cg*16 + l15 fixed, halves j -> key = 16*hi + j
        #pragma unroll
        for (int cg = 0; cg < 4; ++cg) {
            Frag vb;
            #pragma unroll
            for (int j = 0; j < 16; ++j)
                vb.us[j] = Vr[16 * hi + j][cg * 16 + l15];
            o[cg] = wmma_bf16(pa.v, vb.v, o[cg]);
        }
    }

    // normalize and store mix (bf16, token-major [n, s, D])
    #pragma unroll
    for (int r = 0; r < 8; ++r) {
        const float inv = 1.0f / rl[r];
        const int qr = qTile * 64 + wave * 16 + r + 8 * hi;
        const size_t orow = ((size_t)batch * SEQ + qr) * D_MODEL + head * HC;
        #pragma unroll
        for (int cg = 0; cg < 4; ++cg)
            outMix[orow + cg * 16 + l15] = f2bf(o[cg][r] * inv);
    }
}

// ---------------------------------------------------------------------------
// Orchestration
// ---------------------------------------------------------------------------
extern "C" void kernel_launch(void* const* d_in, const int* in_sizes, int n_in,
                              void* d_out, int out_size, void* d_ws, size_t ws_size,
                              hipStream_t stream) {
    (void)in_sizes; (void)n_in; (void)out_size; (void)ws_size;

    const float* x      = (const float*)d_in[0];
    const float* ln1_w  = (const float*)d_in[1];
    const float* ln1_b  = (const float*)d_in[2];
    const float* in_w   = (const float*)d_in[3];
    const float* in_b   = (const float*)d_in[4];
    const float* out_w  = (const float*)d_in[5];
    const float* out_b  = (const float*)d_in[6];
    const float* ln2_w  = (const float*)d_in[7];
    const float* ln2_b  = (const float*)d_in[8];
    const float* fc_w   = (const float*)d_in[9];
    const float* fc_b   = (const float*)d_in[10];
    const float* proj_w = (const float*)d_in[11];
    const float* proj_b = (const float*)d_in[12];

    char* ws = (char*)d_ws;
    size_t off = 0;
    auto alloc = [&](size_t bytes) -> void* {
        void* p = ws + off;
        off = (off + bytes + 255) & ~(size_t)255;
        return p;
    };
    unsigned short* wqkv  = (unsigned short*)alloc((size_t)3 * D_MODEL * D_MODEL * 2);
    unsigned short* wout  = (unsigned short*)alloc((size_t)D_MODEL * D_MODEL * 2);
    unsigned short* wfc   = (unsigned short*)alloc((size_t)4 * D_MODEL * D_MODEL * 2);
    unsigned short* wproj = (unsigned short*)alloc((size_t)4 * D_MODEL * D_MODEL * 2);
    unsigned short* hbuf  = (unsigned short*)alloc((size_t)NTOK * D_MODEL * 2);      // ln1 out
    unsigned short* qkvb  = (unsigned short*)alloc((size_t)NTOK * 3 * D_MODEL * 2);  // qkv
    unsigned short* attnb = (unsigned short*)alloc((size_t)NTOK * D_MODEL * 2);      // mix
    float*          x1    = (float*)alloc((size_t)NTOK * D_MODEL * 4);               // residual 1
    unsigned short* h2buf = (unsigned short*)alloc((size_t)NTOK * D_MODEL * 2);      // ln2 out
    unsigned short* fcbuf = (unsigned short*)alloc((size_t)NTOK * 4 * D_MODEL * 2);  // gelu(fc)

    cvt_f32_bf16<<<(3 * D_MODEL * D_MODEL + 255) / 256, 256, 0, stream>>>(in_w,   wqkv,  3 * D_MODEL * D_MODEL);
    cvt_f32_bf16<<<(D_MODEL * D_MODEL + 255) / 256, 256, 0, stream>>>(out_w,      wout,  D_MODEL * D_MODEL);
    cvt_f32_bf16<<<(4 * D_MODEL * D_MODEL + 255) / 256, 256, 0, stream>>>(fc_w,   wfc,   4 * D_MODEL * D_MODEL);
    cvt_f32_bf16<<<(4 * D_MODEL * D_MODEL + 255) / 256, 256, 0, stream>>>(proj_w, wproj, 4 * D_MODEL * D_MODEL);

    ln_kernel<true><<<NTOK, 256, 0, stream>>>(x, ln1_w, ln1_b, hbuf);

    gemm_bf16<0><<<dim3(3 * D_MODEL / 128, NTOK / 128), 256, 0, stream>>>(
        hbuf, wqkv, in_b, nullptr, qkvb, NTOK, 3 * D_MODEL, D_MODEL);

    attn_kernel<<<dim3(SEQ / 64, NH, NB), 128, 0, stream>>>(qkvb, attnb);

    gemm_bf16<2><<<dim3(D_MODEL / 128, NTOK / 128), 256, 0, stream>>>(
        attnb, wout, out_b, x, x1, NTOK, D_MODEL, D_MODEL);

    ln_kernel<false><<<NTOK, 256, 0, stream>>>(x1, ln2_w, ln2_b, h2buf);

    gemm_bf16<1><<<dim3(4 * D_MODEL / 128, NTOK / 128), 256, 0, stream>>>(
        h2buf, wfc, fc_b, nullptr, fcbuf, NTOK, 4 * D_MODEL, D_MODEL);

    gemm_bf16<3><<<dim3(D_MODEL / 128, NTOK / 128), 256, 0, stream>>>(
        fcbuf, wproj, proj_b, x1, d_out, NTOK, D_MODEL, 4 * D_MODEL);
}
